// MultiHeadSelfAttention_78907139162304
// MI455X (gfx1250) — matrix operand order repaired and checked by reading, not compiled, not run on hardware
//
#include <hip/hip_runtime.h>
#include <hip/hip_bf16.h>

typedef __bf16 bf16;
typedef __attribute__((ext_vector_type(16))) __bf16 bf16x16;
typedef __attribute__((ext_vector_type(8)))  __bf16 bf16x8;
typedef __attribute__((ext_vector_type(8)))  float  f32x8;

typedef __attribute__((ext_vector_type(4))) unsigned int u32x4;
typedef __attribute__((ext_vector_type(8))) int          i32x8;
typedef __attribute__((ext_vector_type(4))) int          i32x4;

union BF16Frag { bf16x16 v; bf16x8 h[2]; };

#define D_MODEL 1024
#define NH      16
#define HDIM    64
#define SEQ     2048
#define BATCH   4
#define ROWS    (BATCH * SEQ)   // 8192

static __device__ __forceinline__ f32x8 zero8() {
    f32x8 z;
#pragma unroll
    for (int e = 0; e < 8; ++e) z[e] = 0.0f;
    return z;
}

// LDS byte offset of a shared-memory pointer (low 32 bits of the generic addr)
static __device__ __forceinline__ unsigned lds_off(const void* p) {
    return (unsigned)(unsigned long long)(size_t)p;
}

// ---------------------------------------------------------------------------
// TDM: issue a 2D bf16 tile load (tile_d1 rows x tile_d0 elems, row stride
// stride0 elems) from global to LDS, packed row-major. D# layout per
// CDNA5 ISA (group0 + group1; groups 2/3 zero for 2D).
// Wave-level op; tracked by TENSORcnt. 6-arg builtin on this toolchain.
// ---------------------------------------------------------------------------
static __device__ __forceinline__ void tdm_load_2d_bf16(
    const bf16* gptr, unsigned ldsoff,
    unsigned tensor_d0, unsigned tensor_d1,
    unsigned tile_d0, unsigned tile_d1, unsigned long long stride0)
{
    unsigned long long ga = (unsigned long long)(size_t)gptr;
    u32x4 g0;
    g0[0] = 1u;                                   // count=1, user mode
    g0[1] = ldsoff;                               // lds_addr
    g0[2] = (unsigned)(ga & 0xffffffffull);       // global_addr[31:0]
    g0[3] = (unsigned)((ga >> 32) & 0x1ffffffull) // global_addr[56:32]
          | (2u << 30);                           // type = 2 ("image")
    i32x8 g1;
    g1[0] = (int)(1u << 16);                      // data_size = 1 -> 2 bytes
    g1[1] = (int)((tensor_d0 & 0xffffu) << 16);   // tensor_dim0[15:0]
    g1[2] = (int)((tensor_d0 >> 16) | ((tensor_d1 & 0xffffu) << 16));
    g1[3] = (int)((tensor_d1 >> 16) | (tile_d0 << 16));
    g1[4] = (int)(tile_d1 & 0xffffu);             // tile_dim1; tile_dim2 = 0
    g1[5] = (int)(unsigned)(stride0 & 0xffffffffull);
    g1[6] = (int)(unsigned)((stride0 >> 32) & 0xffffull);
    g1[7] = 0;
    i32x4 g2 = {0, 0, 0, 0};
    i32x4 g3 = {0, 0, 0, 0};
    i32x8 g4 = {0, 0, 0, 0, 0, 0, 0, 0};
    __builtin_amdgcn_tensor_load_to_lds(g0, g1, g2, g3, g4, 0);
}

// ---------------------------------------------------------------------------
// fp32 -> bf16 conversion
// ---------------------------------------------------------------------------
__global__ void cvt_f32_bf16(const float* __restrict__ in, bf16* __restrict__ out, int n) {
    int i = blockIdx.x * blockDim.x + threadIdx.x;
    if (i < n) out[i] = (bf16)in[i];
}

// ---------------------------------------------------------------------------
// fp32 [K][N] -> bf16 transposed [N][K], 32x32 LDS tiles, block (32,8)
// ---------------------------------------------------------------------------
__global__ void __launch_bounds__(256)
cvt_transpose_bf16(const float* __restrict__ in, bf16* __restrict__ out, int K, int N) {
    __shared__ float tile[32][33];
    const int k0 = blockIdx.y * 32;
    const int n0 = blockIdx.x * 32;
    const int tx = threadIdx.x;   // 0..31
    const int ty = threadIdx.y;   // 0..7
#pragma unroll
    for (int i = 0; i < 32; i += 8)
        tile[ty + i][tx] = in[(size_t)(k0 + ty + i) * N + n0 + tx];
    __syncthreads();
#pragma unroll
    for (int i = 0; i < 32; i += 8)
        out[(size_t)(n0 + ty + i) * K + k0 + tx] = (bf16)tile[tx][ty + i];
}

// ---------------------------------------------------------------------------
// Tiled bf16 GEMM with TDM tile staging:
//   C[M x N] = A[M x K] * Bt[N x K]^T + bias
// A row-major, Bt row-major (pre-transposed weights). Block tile 128x128,
// 256 threads (8 waves), wave tile 32x64 (2x4 WMMA tiles). Tiles are DMA'd
// to LDS by wave 0 via tensor_load_to_lds, double-buffered on TENSORcnt.
// ---------------------------------------------------------------------------
template <bool OUT_BF16>
__global__ void __launch_bounds__(256)
gemm_bf16(const bf16* __restrict__ A, const bf16* __restrict__ Bt,
          const float* __restrict__ bias, void* __restrict__ Cout,
          int M, int N, int K) {
    __shared__ bf16 Alds[2][128][32];
    __shared__ bf16 Blds[2][128][32];   // [n][k]

    const int tid  = threadIdx.x;
    const int lane = tid & 31;
    const int wave = tid >> 5;
    const int hf   = (lane >> 4) & 1;
    const int ln   = lane & 15;
    const int wm   = wave >> 1;      // 0..3 -> M
    const int wn   = wave & 1;       // 0..1 -> N
    const int m0   = blockIdx.y * 128;
    const int n0   = blockIdx.x * 128;

    f32x8 acc[2][4];
#pragma unroll
    for (int i = 0; i < 2; ++i)
#pragma unroll
        for (int j = 0; j < 4; ++j) acc[i][j] = zero8();

    const int nsteps = K >> 5;   // K/32

    if (wave == 0) {
        tdm_load_2d_bf16(A  + (size_t)m0 * K, lds_off(&Alds[0][0][0]),
                         (unsigned)K, (unsigned)(M - m0), 32, 128, (unsigned long long)K);
        tdm_load_2d_bf16(Bt + (size_t)n0 * K, lds_off(&Blds[0][0][0]),
                         (unsigned)K, (unsigned)(N - n0), 32, 128, (unsigned long long)K);
    }

    for (int s = 0; s < nsteps; ++s) {
        const int buf = s & 1;
        if (wave == 0) {
            if (s + 1 < nsteps) {
                const int k0 = (s + 1) << 5;
                tdm_load_2d_bf16(A  + (size_t)m0 * K + k0, lds_off(&Alds[buf ^ 1][0][0]),
                                 (unsigned)(K - k0), (unsigned)(M - m0), 32, 128,
                                 (unsigned long long)K);
                tdm_load_2d_bf16(Bt + (size_t)n0 * K + k0, lds_off(&Blds[buf ^ 1][0][0]),
                                 (unsigned)(K - k0), (unsigned)(N - n0), 32, 128,
                                 (unsigned long long)K);
                __builtin_amdgcn_s_wait_tensorcnt(2);   // current buf's pair done
            } else {
                __builtin_amdgcn_s_wait_tensorcnt(0);
            }
        }
        __syncthreads();   // publish TDM-written LDS to all waves

        BF16Frag af[2], bf[4];
#pragma unroll
        for (int mt = 0; mt < 2; ++mt) {
            int m = wm * 32 + mt * 16 + ln;
            af[mt].h[0] = *(const bf16x8*)(&Alds[buf][m][hf * 8]);
            af[mt].h[1] = *(const bf16x8*)(&Alds[buf][m][16 + hf * 8]);
        }
#pragma unroll
        for (int nt = 0; nt < 4; ++nt) {
            int n = wn * 64 + nt * 16 + ln;
            bf[nt].h[0] = *(const bf16x8*)(&Blds[buf][n][hf * 8]);
            bf[nt].h[1] = *(const bf16x8*)(&Blds[buf][n][16 + hf * 8]);
        }
#pragma unroll
        for (int mt = 0; mt < 2; ++mt)
#pragma unroll
            for (int nt = 0; nt < 4; ++nt)
                acc[mt][nt] = __builtin_amdgcn_wmma_f32_16x16x32_bf16(
                    false, af[mt].v, false, bf[nt].v, (short)0, acc[mt][nt], false, false);

        __syncthreads();   // all reads of buf done before it is reloaded
    }

#pragma unroll
    for (int mt = 0; mt < 2; ++mt)
#pragma unroll
        for (int nt = 0; nt < 4; ++nt) {
            int n    = n0 + wn * 64 + nt * 16 + ln;
            float bn = bias ? bias[n] : 0.0f;
#pragma unroll
            for (int r = 0; r < 8; ++r) {
                int m   = m0 + wm * 32 + mt * 16 + r + 8 * hf;
                float v = acc[mt][nt][r] + bn;
                if (OUT_BF16) ((bf16*)Cout)[(size_t)m * N + n] = (bf16)v;
                else          ((float*)Cout)[(size_t)m * N + n] = v;
            }
        }
}

// ---------------------------------------------------------------------------
// Flash attention: one block per (64 queries, head, batch); 4 waves,
// each wave owns 16 query rows. K/V blocks of 64 staged in LDS.
// qkv layout: (B, N, 3, H, HD) bf16 -> row stride 3*D_MODEL.
// ctx layout: (B, N, H*HD) bf16.
// ---------------------------------------------------------------------------
__global__ void __launch_bounds__(128)
attn_kernel(const bf16* __restrict__ qkv, bf16* __restrict__ ctx) {
    __shared__ bf16 Qs[64][64];
    __shared__ bf16 Ks[64][64];
    __shared__ bf16 Vs[64][64];   // transposed: [d][key]
    __shared__ bf16 Ps[64][64];

    const int tid  = threadIdx.x;
    const int lane = tid & 31;
    const int wave = tid >> 5;    // 0..3
    const int hf   = (lane >> 4) & 1;
    const int ln   = lane & 15;
    const int q0   = blockIdx.x * 64;
    const int h    = blockIdx.y;
    const int b    = blockIdx.z;

    const size_t rowstride = 3 * D_MODEL;
    const size_t baseQ = (size_t)b * SEQ * rowstride + (size_t)h * HDIM;
    const size_t baseK = baseQ + D_MODEL;
    const size_t baseV = baseQ + 2 * D_MODEL;

    {
        int row = tid >> 1;
        int seg = tid & 1;
        const bf16* src = qkv + baseQ + (size_t)(q0 + row) * rowstride + seg * 32;
#pragma unroll
        for (int i = 0; i < 4; ++i)
            *(uint4*)(&Qs[row][seg * 32 + i * 8]) = *(const uint4*)(src + i * 8);
    }

    f32x8 acco[4];
#pragma unroll
    for (int dt = 0; dt < 4; ++dt) acco[dt] = zero8();
    float m_i[8], l_i[8];
#pragma unroll
    for (int r = 0; r < 8; ++r) { m_i[r] = -1e30f; l_i[r] = 0.0f; }

    for (int kb = 0; kb < SEQ / 64; ++kb) {
        const int key0 = kb * 64;
        __syncthreads();
        {
            int row = tid >> 1, seg = tid & 1;
            const bf16* src = qkv + baseK + (size_t)(key0 + row) * rowstride + seg * 32;
#pragma unroll
            for (int i = 0; i < 4; ++i)
                *(uint4*)(&Ks[row][seg * 32 + i * 8]) = *(const uint4*)(src + i * 8);
        }
        {
            int d  = tid & 63;
            int kh = tid >> 6;   // 0..1
            bf16 tmp[32] __attribute__((aligned(16)));
#pragma unroll
            for (int j = 0; j < 32; ++j)
                tmp[j] = qkv[baseV + (size_t)(key0 + kh * 32 + j) * rowstride + d];
#pragma unroll
            for (int i = 0; i < 4; ++i)
                *(bf16x8*)(&Vs[d][kh * 32 + i * 8]) = *(const bf16x8*)(&tmp[i * 8]);
        }
        __syncthreads();

        // S = Q * K^T
        f32x8 accs[4];
#pragma unroll
        for (int nt = 0; nt < 4; ++nt) accs[nt] = zero8();
#pragma unroll
        for (int kk = 0; kk < 2; ++kk) {
            BF16Frag a;
            int m = wave * 16 + ln;
            a.h[0] = *(const bf16x8*)(&Qs[m][kk * 32 + hf * 8]);
            a.h[1] = *(const bf16x8*)(&Qs[m][kk * 32 + 16 + hf * 8]);
#pragma unroll
            for (int nt = 0; nt < 4; ++nt) {
                BF16Frag bm;
                int n = nt * 16 + ln;
                bm.h[0] = *(const bf16x8*)(&Ks[n][kk * 32 + hf * 8]);
                bm.h[1] = *(const bf16x8*)(&Ks[n][kk * 32 + 16 + hf * 8]);
                accs[nt] = __builtin_amdgcn_wmma_f32_16x16x32_bf16(
                    false, a.v, false, bm.v, (short)0, accs[nt], false, false);
            }
        }

        const float scale = 0.125f;   // 1/sqrt(64)
#pragma unroll
        for (int r = 0; r < 8; ++r) {
            float s0 = accs[0][r] * scale, s1 = accs[1][r] * scale;
            float s2 = accs[2][r] * scale, s3 = accs[3][r] * scale;
            float rowmax = fmaxf(fmaxf(s0, s1), fmaxf(s2, s3));
#pragma unroll
            for (int off = 1; off < 16; off <<= 1)
                rowmax = fmaxf(rowmax, __shfl_xor(rowmax, off, 32));
            float mnew  = fmaxf(m_i[r], rowmax);
            float alpha = __expf(m_i[r] - mnew);
            float p0 = __expf(s0 - mnew), p1 = __expf(s1 - mnew);
            float p2 = __expf(s2 - mnew), p3 = __expf(s3 - mnew);
            float rs = p0 + p1 + p2 + p3;
#pragma unroll
            for (int off = 1; off < 16; off <<= 1)
                rs += __shfl_xor(rs, off, 32);
            l_i[r] = l_i[r] * alpha + rs;
            m_i[r] = mnew;
#pragma unroll
            for (int dt = 0; dt < 4; ++dt) acco[dt][r] *= alpha;
            int m = wave * 16 + r + 8 * hf;
            Ps[m][0 * 16 + ln] = (bf16)p0;
            Ps[m][1 * 16 + ln] = (bf16)p1;
            Ps[m][2 * 16 + ln] = (bf16)p2;
            Ps[m][3 * 16 + ln] = (bf16)p3;
        }
        __syncthreads();

        // O += P * V
#pragma unroll
        for (int kk = 0; kk < 2; ++kk) {
            BF16Frag a;
            int m = wave * 16 + ln;
            a.h[0] = *(const bf16x8*)(&Ps[m][kk * 32 + hf * 8]);
            a.h[1] = *(const bf16x8*)(&Ps[m][kk * 32 + 16 + hf * 8]);
#pragma unroll
            for (int dt = 0; dt < 4; ++dt) {
                BF16Frag bm;
                int n = dt * 16 + ln;
                bm.h[0] = *(const bf16x8*)(&Vs[n][kk * 32 + hf * 8]);
                bm.h[1] = *(const bf16x8*)(&Vs[n][kk * 32 + 16 + hf * 8]);
                acco[dt] = __builtin_amdgcn_wmma_f32_16x16x32_bf16(
                    false, a.v, false, bm.v, (short)0, acco[dt], false, false);
            }
        }
    }

#pragma unroll
    for (int r = 0; r < 8; ++r) {
        float inv = 1.0f / l_i[r];
        int q = q0 + wave * 16 + r + 8 * hf;
        size_t base = ((size_t)b * SEQ + q) * D_MODEL + (size_t)h * HDIM;
#pragma unroll
        for (int dt = 0; dt < 4; ++dt)
            ctx[base + dt * 16 + ln] = (bf16)(acco[dt][r] * inv);
    }
}

// ---------------------------------------------------------------------------
extern "C" void kernel_launch(void* const* d_in, const int* in_sizes, int n_in,
                              void* d_out, int out_size, void* d_ws, size_t ws_size,
                              hipStream_t stream) {
    const float* x     = (const float*)d_in[0];
    const float* w_qkv = (const float*)d_in[1];
    const float* b_qkv = (const float*)d_in[2];
    const float* w_out = (const float*)d_in[3];
    const float* b_out = (const float*)d_in[4];
    float* out = (float*)d_out;

    char* ws = (char*)d_ws;
    bf16* x_b    = (bf16*)ws; ws += (size_t)ROWS * D_MODEL * 2;
    bf16* wqkvT  = (bf16*)ws; ws += (size_t)D_MODEL * 3 * D_MODEL * 2;  // [3D][D]
    bf16* woutT  = (bf16*)ws; ws += (size_t)D_MODEL * D_MODEL * 2;      // [D][D]
    bf16* qkv_b  = (bf16*)ws; ws += (size_t)ROWS * 3 * D_MODEL * 2;
    bf16* ctx_b  = (bf16*)ws;

    int n1 = ROWS * D_MODEL;
    cvt_f32_bf16<<<(n1 + 255) / 256, 256, 0, stream>>>(x, x_b, n1);

    // weights: convert + transpose to [N][K] bf16
    dim3 tb(32, 8);
    dim3 tg1(3 * D_MODEL / 32, D_MODEL / 32);
    cvt_transpose_bf16<<<tg1, tb, 0, stream>>>(w_qkv, wqkvT, D_MODEL, 3 * D_MODEL);
    dim3 tg2(D_MODEL / 32, D_MODEL / 32);
    cvt_transpose_bf16<<<tg2, tb, 0, stream>>>(w_out, woutT, D_MODEL, D_MODEL);

    // QKV projection: (8192 x 1024) @ (1024 x 3072) + b_qkv -> bf16
    dim3 g1(3 * D_MODEL / 128, ROWS / 128);
    gemm_bf16<true><<<g1, 256, 0, stream>>>(x_b, wqkvT, b_qkv, (void*)qkv_b,
                                            ROWS, 3 * D_MODEL, D_MODEL);

    // attention
    dim3 g2(SEQ / 64, NH, BATCH);
    attn_kernel<<<g2, 128, 0, stream>>>(qkv_b, ctx_b);

    // output projection: (8192 x 1024) @ (1024 x 1024) + b_out -> fp32 d_out
    dim3 g3(D_MODEL / 128, ROWS / 128);
    gemm_bf16<false><<<g3, 256, 0, stream>>>(ctx_b, woutT, b_out, (void*)out,
                                             ROWS, D_MODEL, D_MODEL);
}
